// rWindowAttention_80865644249597
// MI455X (gfx1250) — matrix operand order repaired and checked by reading, not compiled, not run on hardware
//
#include <hip/hip_runtime.h>
#include <hip/hip_bf16.h>

typedef __bf16 bf16;
typedef __attribute__((ext_vector_type(16))) __bf16 v16bf;
typedef __attribute__((ext_vector_type(8)))  float  v8f;
typedef unsigned int v4u __attribute__((ext_vector_type(4)));
typedef int          v8i __attribute__((ext_vector_type(8)));
typedef int          v4i __attribute__((ext_vector_type(4)));

#define WMMA_BF16(a, b, c) \
  __builtin_amdgcn_wmma_f32_16x16x32_bf16(false, (a), false, (b), (short)0, (c), false, false)

#if defined(__has_builtin)
#if __has_builtin(__builtin_amdgcn_tensor_load_to_lds)
#define HAVE_TDM 1
#endif
#endif

// Problem constants
#define BATCH   1024
#define NTOK    256
#define CH      128
#define NH      4
#define DH      32
#define NSLICE  4
#define QSLICE  64
#define TOTROWS (BATCH * NTOK)   // 262144

#ifdef HAVE_TDM
// Issue a 1-D TDM copy of `elems` bf16 from `gsrc` to LDS offset `lds_off`,
// inserting 4 dwords of padding every 16 dwords (row stride 32 -> 40 elems).
// Must be executed by exactly one wave.
__device__ __forceinline__ void tdm_load_padded(unsigned lds_off,
                                                const bf16* gsrc,
                                                unsigned elems) {
  unsigned long long ga = (unsigned long long)gsrc;
  v4u g0;
  g0[0] = 1u;                                     // count=1, user descriptor
  g0[1] = lds_off;                                // LDS byte address
  g0[2] = (unsigned)(ga & 0xffffffffu);           // global_addr[31:0]
  g0[3] = (unsigned)((ga >> 32) & 0x1ffffffu) | (2u << 30);  // addr[56:32] | type=2
  v8i g1;
  // data_size=2B(1)<<16 | pad_enable<<20 | pad_interval=16dw(3)<<22 | pad_amount=4dw(3)<<25
  g1[0] = (int)((1u << 16) | (1u << 20) | (3u << 22) | (3u << 25));
  g1[1] = (int)((elems & 0xffffu) << 16);         // tensor_dim0[15:0]
  g1[2] = (int)(((elems >> 16) & 0xffffu) | (1u << 16)); // tensor_dim0[31:16] | tensor_dim1=1
  g1[3] = (int)((elems & 0xffffu) << 16);         // tensor_dim1[31:16]=0 | tile_dim0
  g1[4] = 0;                                      // tile_dim1=0, tile_dim2=0
  g1[5] = (int)elems;                             // tensor_dim0_stride[31:0]
  g1[6] = 0;
  g1[7] = 0;
  v4i gz = {0, 0, 0, 0};
#if __clang_major__ >= 23
  v8i gz8 = {0, 0, 0, 0, 0, 0, 0, 0};
  __builtin_amdgcn_tensor_load_to_lds(g0, g1, gz, gz, gz8, 0);
#else
  __builtin_amdgcn_tensor_load_to_lds(g0, g1, gz, gz, 0);
#endif
}
#endif

// ---------------------------------------------------------------------------
// Kernel 0: convert weights f32 -> bf16
// ---------------------------------------------------------------------------
__global__ __launch_bounds__(256) void k_cvt_w(const float* __restrict__ qw,
                                               const float* __restrict__ kvw,
                                               const float* __restrict__ pw,
                                               bf16* __restrict__ wqb,
                                               bf16* __restrict__ wkvb,
                                               bf16* __restrict__ wpb) {
  int i = blockIdx.x * 256 + threadIdx.x;
  if (i < CH * CH)     wqb[i]  = (bf16)qw[i];
  if (i < CH * 2 * CH) wkvb[i] = (bf16)kvw[i];
  if (i < CH * CH)     wpb[i]  = (bf16)pw[i];
}

// ---------------------------------------------------------------------------
// Kernel 1: fused QKV projection.  X[262144,128] @ [Wq | Wkv] -> q,k,v (bf16)
// q pre-scaled by Dh^-0.5.  Outputs in [B,H,N,Dh] layout.
// Block: 256 thr (8 waves), 128 rows of X. Wave w owns row-tile w.
// Col-tile loops split per destination (q / k / v) -> straight-line stores.
// ---------------------------------------------------------------------------
__global__ __launch_bounds__(256) void k_qkv(const float* __restrict__ x,
                                             const bf16* __restrict__ wq,
                                             const bf16* __restrict__ wkv,
                                             bf16* __restrict__ qb,
                                             bf16* __restrict__ kb,
                                             bf16* __restrict__ vb) {
  __shared__ bf16 Xs[128][136];  // padded stride vs 64-bank LDS
  const int t = threadIdx.x;
  const long rowBase = (long)blockIdx.x * 128;

  for (int i = 0; i < 64; ++i) {
    int flat = i * 256 + t;
    int r = flat >> 7, c = flat & 127;
    Xs[r][c] = (bf16)x[(rowBase + r) * CH + c];
  }
  __syncthreads();

  const int lane = t & 31, w = t >> 5;
  const int l = lane & 15, g = lane >> 4;
  const float qscale = 0.17677669529663687f;  // 32^-0.5
  const int arow = w * 16 + l;

  // ---- q tiles
  for (int ct = 0; ct < 8; ++ct) {
    v8f acc = {};
    for (int kt = 0; kt < 4; ++kt) {
      v16bf a, bb;
      const int kb0 = kt * 32 + g * 8;
#pragma unroll
      for (int j = 0; j < 8; ++j) {
        a[j]     = Xs[arow][kb0 + j];
        a[8 + j] = Xs[arow][kb0 + 16 + j];
      }
      const bf16* bp = wq + (long)(kt * 32 + g * 8) * CH + ct * 16 + l;
#pragma unroll
      for (int j = 0; j < 8; ++j) { bb[j] = bp[j * CH]; bb[8 + j] = bp[(16 + j) * CH]; }
      acc = WMMA_BF16(a, bb, acc);
    }
    const int col = ct * 16 + l, h = col >> 5, d = col & 31;
#pragma unroll
    for (int r = 0; r < 8; ++r) {
      long gr = rowBase + w * 16 + r + 8 * g;
      long b = gr >> 8, n = gr & 255;
      qb[(((b * NH + h) * NTOK) + n) * DH + d] = (bf16)(acc[r] * qscale);
    }
  }
  // ---- k tiles (kv cols 0..127)
  for (int ct = 0; ct < 8; ++ct) {
    v8f acc = {};
    for (int kt = 0; kt < 4; ++kt) {
      v16bf a, bb;
      const int kb0 = kt * 32 + g * 8;
#pragma unroll
      for (int j = 0; j < 8; ++j) {
        a[j]     = Xs[arow][kb0 + j];
        a[8 + j] = Xs[arow][kb0 + 16 + j];
      }
      const bf16* bp = wkv + (long)(kt * 32 + g * 8) * (2 * CH) + ct * 16 + l;
#pragma unroll
      for (int j = 0; j < 8; ++j) { bb[j] = bp[j * (2 * CH)]; bb[8 + j] = bp[(16 + j) * (2 * CH)]; }
      acc = WMMA_BF16(a, bb, acc);
    }
    const int col = ct * 16 + l, h = col >> 5, d = col & 31;
#pragma unroll
    for (int r = 0; r < 8; ++r) {
      long gr = rowBase + w * 16 + r + 8 * g;
      long b = gr >> 8, n = gr & 255;
      kb[(((b * NH + h) * NTOK) + n) * DH + d] = (bf16)acc[r];
    }
  }
  // ---- v tiles (kv cols 128..255)
  for (int ct = 0; ct < 8; ++ct) {
    v8f acc = {};
    for (int kt = 0; kt < 4; ++kt) {
      v16bf a, bb;
      const int kb0 = kt * 32 + g * 8;
#pragma unroll
      for (int j = 0; j < 8; ++j) {
        a[j]     = Xs[arow][kb0 + j];
        a[8 + j] = Xs[arow][kb0 + 16 + j];
      }
      const bf16* bp = wkv + (long)(kt * 32 + g * 8) * (2 * CH) + CH + ct * 16 + l;
#pragma unroll
      for (int j = 0; j < 8; ++j) { bb[j] = bp[j * (2 * CH)]; bb[8 + j] = bp[(16 + j) * (2 * CH)]; }
      acc = WMMA_BF16(a, bb, acc);
    }
    const int col = ct * 16 + l, h = col >> 5, d = col & 31;
#pragma unroll
    for (int r = 0; r < 8; ++r) {
      long gr = rowBase + w * 16 + r + 8 * g;
      long b = gr >> 8, n = gr & 255;
      vb[(((b * NH + h) * NTOK) + n) * DH + d] = (bf16)acc[r];
    }
  }
}

// ---------------------------------------------------------------------------
// Kernel 2: attention for one (window b, head h, slice s).
// Q (64x32) and K (256x32) staged into LDS via the Tensor Data Mover
// (padding 32->40 elem stride done by the TDM pad feature).
// S = Q @ K^T + bias, same-slice keys masked to -inf, softmax over 256,
// O = P(64x256) @ V(256x32) -> ab (bf16, [B,N,C]).
// ---------------------------------------------------------------------------
__global__ __launch_bounds__(256) void k_attn(const bf16* __restrict__ qb,
                                              const bf16* __restrict__ kb,
                                              const bf16* __restrict__ vb,
                                              const float* __restrict__ rpb,
                                              const int* __restrict__ rel,
                                              bf16* __restrict__ ab) {
  __shared__ bf16  Qs[QSLICE][40];
  __shared__ bf16  Ks[NTOK][40];
  __shared__ float Sf[QSLICE][260];
  __shared__ bf16  Pb[QSLICE][264];

  const int t = threadIdx.x;
  const int idx = blockIdx.x;
  const int b = idx >> 4;
  const int h = (idx >> 2) & 3;
  const int s = idx & 3;

  const bf16* qrow = qb + ((((long)b * NH + h) * NTOK) + s * QSLICE) * DH;
  const bf16* krow = kb + (((long)b * NH + h) * NTOK) * DH;
  const bf16* vrow = vb + (((long)b * NH + h) * NTOK) * DH;

#ifdef HAVE_TDM
  if (t < 32) {  // wave 0 issues both DMA descriptors
    tdm_load_padded((unsigned)(unsigned long long)&Qs[0][0], qrow, QSLICE * DH);
    tdm_load_padded((unsigned)(unsigned long long)&Ks[0][0], krow, NTOK * DH);
    __builtin_amdgcn_s_wait_tensorcnt(0);
  }
#else
  for (int i = 0; i < 8; ++i) {  // 64x32 Q
    int flat = i * 256 + t;
    int r = flat >> 5, c = flat & 31;
    Qs[r][c] = qrow[r * DH + c];
  }
  for (int i = 0; i < 32; ++i) {  // 256x32 K
    int flat = i * 256 + t;
    int r = flat >> 5, c = flat & 31;
    Ks[r][c] = krow[r * DH + c];
  }
#endif
  __syncthreads();

  const int lane = t & 31, w = t >> 5;
  const int l = lane & 15, g = lane >> 4;

  // ---- scores: 64x256 = 4x16 tiles of 16x16, 8 tiles per wave, 1 WMMA each
  for (int tt = 0; tt < 8; ++tt) {
    int tileIdx = w * 8 + tt;
    int rt = tileIdx >> 4, ct = tileIdx & 15;
    v16bf a, bb;
    const int arow = rt * 16 + l;
    const int kb0 = g * 8;
#pragma unroll
    for (int j = 0; j < 8; ++j) {
      a[j]     = Qs[arow][kb0 + j];
      a[8 + j] = Qs[arow][kb0 + 16 + j];
    }
#pragma unroll
    for (int j = 0; j < 8; ++j) { bb[j] = Ks[ct * 16 + l][kb0 + j]; bb[8 + j] = Ks[ct * 16 + l][kb0 + 16 + j]; }  // K^T column
    v8f acc = {};
    acc = WMMA_BF16(a, bb, acc);

    const int kc = ct * 16 + l;                 // key token 0..255
    const bool masked = ((kc >> 6) == s);       // same time-slice -> excluded
#pragma unroll
    for (int r = 0; r < 8; ++r) {
      int M = rt * 16 + r + 8 * g;
      int qg = s * QSLICE + M;
      float val = acc[r] + rpb[rel[qg * NTOK + kc] * NH + h];
      Sf[M][kc] = masked ? -1e30f : val;
    }
  }
  __syncthreads();

  // ---- softmax: 4 lanes per row, 64 cols each, shfl-combine
  {
    const int row = t >> 2, base = (t & 3) * 64;
    float m = -1e30f;
    for (int j = 0; j < 64; ++j) m = fmaxf(m, Sf[row][base + j]);
    m = fmaxf(m, __shfl_xor(m, 1));
    m = fmaxf(m, __shfl_xor(m, 2));
    float sum = 0.f;
    for (int j = 0; j < 64; ++j) {
      float e = __expf(Sf[row][base + j] - m);
      Sf[row][base + j] = e;
      sum += e;
    }
    sum += __shfl_xor(sum, 1);
    sum += __shfl_xor(sum, 2);
    float inv = 1.0f / sum;
    for (int j = 0; j < 64; ++j) Pb[row][base + j] = (bf16)(Sf[row][base + j] * inv);
  }
  __syncthreads();

  // ---- O = P @ V : 64x32 = 8 tiles, one per wave, K=256 -> 8 chained WMMAs
  const int rt = w >> 1, ctO = w & 1;
  v8f acc = {};
  for (int kt = 0; kt < 8; ++kt) {
    v16bf a, bb;
    const int arow = rt * 16 + l;
    const int kb0 = kt * 32 + g * 8;
#pragma unroll
    for (int j = 0; j < 8; ++j) {
      a[j]     = Pb[arow][kb0 + j];
      a[8 + j] = Pb[arow][kb0 + 16 + j];
    }
    const bf16* vp = vrow + (long)(kt * 32 + g * 8) * DH + ctO * 16 + l;
#pragma unroll
    for (int j = 0; j < 8; ++j) { bb[j] = vp[j * DH]; bb[8 + j] = vp[(16 + j) * DH]; }
    acc = WMMA_BF16(a, bb, acc);
  }
#pragma unroll
  for (int r = 0; r < 8; ++r) {
    int M = rt * 16 + r + 8 * g;
    long gr = (long)b * NTOK + s * QSLICE + M;
    int col = h * DH + ctO * 16 + l;
    ab[gr * CH + col] = (bf16)acc[r];
  }
}

// ---------------------------------------------------------------------------
// Kernel 3: output projection  out = ab @ proj_w + proj_b  (f32 out)
// ---------------------------------------------------------------------------
__global__ __launch_bounds__(256) void k_proj(const bf16* __restrict__ ab,
                                              const bf16* __restrict__ wp,
                                              const float* __restrict__ bias,
                                              float* __restrict__ out) {
  __shared__ bf16 Xs[128][136];
  const int t = threadIdx.x;
  const long rowBase = (long)blockIdx.x * 128;

  for (int i = 0; i < 64; ++i) {
    int flat = i * 256 + t;
    int r = flat >> 7, c = flat & 127;
    Xs[r][c] = ab[(rowBase + r) * CH + c];
  }
  __syncthreads();

  const int lane = t & 31, w = t >> 5;
  const int l = lane & 15, g = lane >> 4;

  for (int ct = 0; ct < 8; ++ct) {
    v8f acc = {};
    for (int kt = 0; kt < 4; ++kt) {
      v16bf a, bb;
      const int arow = w * 16 + l;
      const int kb0 = kt * 32 + g * 8;
#pragma unroll
      for (int j = 0; j < 8; ++j) {
        a[j]     = Xs[arow][kb0 + j];
        a[8 + j] = Xs[arow][kb0 + 16 + j];
      }
      const bf16* bp = wp + (long)(kt * 32 + g * 8) * CH + ct * 16 + l;
#pragma unroll
      for (int j = 0; j < 8; ++j) { bb[j] = bp[j * CH]; bb[8 + j] = bp[(16 + j) * CH]; }
      acc = WMMA_BF16(a, bb, acc);
    }
    float bcol = bias[ct * 16 + l];
#pragma unroll
    for (int r = 0; r < 8; ++r) {
      long gr = rowBase + w * 16 + r + 8 * g;
      out[gr * CH + ct * 16 + l] = acc[r] + bcol;
    }
  }
}

// ---------------------------------------------------------------------------
extern "C" void kernel_launch(void* const* d_in, const int* in_sizes, int n_in,
                              void* d_out, int out_size, void* d_ws, size_t ws_size,
                              hipStream_t stream) {
  const float* x      = (const float*)d_in[0];
  const float* q_w    = (const float*)d_in[1];
  const float* kv_w   = (const float*)d_in[2];
  const float* proj_w = (const float*)d_in[3];
  const float* proj_b = (const float*)d_in[4];
  const float* rpb    = (const float*)d_in[5];
  const int*   rel    = (const int*)d_in[6];
  float* out = (float*)d_out;

  // workspace layout (bytes)
  char* ws = (char*)d_ws;
  const size_t QKV_ELEMS = (size_t)BATCH * NH * NTOK * DH;  // 33.5M
  bf16* qb   = (bf16*)(ws);
  bf16* kb   = (bf16*)(ws + QKV_ELEMS * 2);
  bf16* vb   = (bf16*)(ws + QKV_ELEMS * 4);
  bf16* ab   = (bf16*)(ws + QKV_ELEMS * 6);
  bf16* wqb  = (bf16*)(ws + QKV_ELEMS * 8);
  bf16* wkvb = (bf16*)(ws + QKV_ELEMS * 8 + 32768);
  bf16* wpb  = (bf16*)(ws + QKV_ELEMS * 8 + 32768 + 65536);

  k_cvt_w<<<128, 256, 0, stream>>>(q_w, kv_w, proj_w, wqb, wkvb, wpb);
  k_qkv<<<TOTROWS / 128, 256, 0, stream>>>(x, wqb, wkvb, qb, kb, vb);
  k_attn<<<BATCH * NH * NSLICE, 256, 0, stream>>>(qb, kb, vb, rpb, rel, ab);
  k_proj<<<TOTROWS / 128, 256, 0, stream>>>(ab, wpb, proj_b, out);
}
